// OrthogonalProjectionAttention_16887811408338
// MI455X (gfx1250) — matrix operand order repaired and checked by reading, not compiled, not run on hardware
//
#include <hip/hip_runtime.h>
#include <hip/hip_bf16.h>

typedef __attribute__((ext_vector_type(16))) _Float16 v16h;
typedef __attribute__((ext_vector_type(8)))  float    v8f;

#define DEV __device__ __forceinline__

static constexpr int C_   = 768;
static constexpr int C3   = 2304;
static constexpr int HH   = 12;
static constexpr int HD   = 64;
static constexpr int NSEQ = 1024;
static constexpr int BB   = 8;
static constexpr int MTOK = BB * NSEQ;   // 8192 token rows
static constexpr int BH   = BB * HH;     // 96 (b,h) pairs

DEV v8f vzero8() { v8f z = {0.f,0.f,0.f,0.f,0.f,0.f,0.f,0.f}; return z; }

DEV v8f wmma16(v16h a, v16h b, v8f c) {
  return __builtin_amdgcn_wmma_f32_16x16x32_f16(false, a, false, b, (short)0, c, false, false);
}

// A fragment: 16x32 f16, M x K, from row-major source with leading dim ld.
// Lanes 0-15: row M=lane, K = 0..15; lanes 16-31: row M=lane-16, K = 16..31.
DEV v16h load_A16x32(const _Float16* p, int ld) {
  int lane = threadIdx.x & 31;
  const _Float16* row = p + (size_t)(lane & 15) * ld;
  int kb = (lane & 16) ? 16 : 0;
  v16h a;
#pragma unroll
  for (int e = 0; e < 16; ++e) a[e] = row[kb + e];
  return a;
}

// B fragment 32x16 (K x N) taken from a K-contiguous source: row (lane&15) selects the
// B-matrix column, 16 contiguous halves along K. Lanes 0-15 hold K=0..15, 16-31 K=16..31.
DEV v16h load_Bt32x16(const _Float16* p, int ld) {
  int lane = threadIdx.x & 31;
  const _Float16* row = p + (size_t)(lane & 15) * ld + ((lane & 16) ? 16 : 0);
  v16h b;
#pragma unroll
  for (int e = 0; e < 16; ++e) b[e] = row[e];
  return b;
}

// Reductions across the 16-lane half holding one C-matrix row (xor 1,2,4,8 stays in-half).
DEV float hmax16(float v) {
#pragma unroll
  for (int d = 1; d < 16; d <<= 1) v = fmaxf(v, __shfl_xor(v, d, 32));
  return v;
}
DEV float hsum16(float v) {
#pragma unroll
  for (int d = 1; d < 16; d <<= 1) v += __shfl_xor(v, d, 32);
  return v;
}

// ---------------------------------------------------------------- convert f32 -> f16
__global__ void opa_cvt_f16(const float* __restrict__ src, _Float16* __restrict__ dst, int n) {
  for (int i = blockIdx.x * blockDim.x + threadIdx.x; i < n; i += gridDim.x * blockDim.x)
    dst[i] = (_Float16)src[i];
}

// ---------------------------------------------------------------- QKV GEMM: Y = X[8192x768] @ W[2304x768]^T
// WG = 8 waves (2x4); wave computes a 32x64 block (2x4 WMMA tiles).
// Epilogue scatters Q,K into [B,H,N,D]; V into TRANSPOSED [B,H,D,N] so the A*V GEMM
// in attention can use contiguous K-major fragment loads.
__global__ __launch_bounds__(256)
void opa_gemm_qkv(const _Float16* __restrict__ X, const _Float16* __restrict__ W,
                  _Float16* __restrict__ Q, _Float16* __restrict__ K, _Float16* __restrict__ Vt) {
  int wid  = threadIdx.x >> 5;
  int lane = threadIdx.x & 31;
  int row0 = blockIdx.y * 64  + (wid >> 2) * 32;
  int col0 = blockIdx.x * 256 + (wid & 3) * 64;
  v8f acc[2][4];
#pragma unroll
  for (int ri = 0; ri < 2; ++ri)
#pragma unroll
    for (int ci = 0; ci < 4; ++ci) acc[ri][ci] = vzero8();
  for (int kk = 0; kk < C_; kk += 32) {
    v16h a0 = load_A16x32(X + (size_t)row0 * C_ + kk, C_);
    v16h a1 = load_A16x32(X + (size_t)(row0 + 16) * C_ + kk, C_);
    v16h b[4];
#pragma unroll
    for (int ci = 0; ci < 4; ++ci)
      b[ci] = load_Bt32x16(W + (size_t)(col0 + ci * 16) * C_ + kk, C_);
#pragma unroll
    for (int ci = 0; ci < 4; ++ci) {
      acc[0][ci] = wmma16(a0, b[ci], acc[0][ci]);
      acc[1][ci] = wmma16(a1, b[ci], acc[1][ci]);
    }
  }
  int colt = lane & 15;
  int rowh = (lane & 16) ? 8 : 0;
#pragma unroll
  for (int ri = 0; ri < 2; ++ri)
#pragma unroll
    for (int ci = 0; ci < 4; ++ci)
#pragma unroll
      for (int g = 0; g < 8; ++g) {
        int m = row0 + ri * 16 + rowh + g;
        int c = col0 + ci * 16 + colt;
        int s = c / C_;
        int rem = c - s * C_;
        int h = rem >> 6, d = rem & 63;
        int b_ = m >> 10, n = m & 1023;
        size_t head = (size_t)(b_ * HH + h);
        _Float16 val = (_Float16)acc[ri][ci][g];
        if (s == 0)      Q[(head * NSEQ + n) * HD + d] = val;
        else if (s == 1) K[(head * NSEQ + n) * HD + d] = val;
        else             Vt[(head * HD + d) * NSEQ + n] = val;  // transposed
      }
}

// ---------------------------------------------------------------- Gram-Schmidt ortho + saliency
__global__ __launch_bounds__(256)
void opa_ortho(_Float16* __restrict__ Qid, const _Float16* __restrict__ Qcl,
               const float* __restrict__ orth_scale, float* __restrict__ sal) {
  int t = blockIdx.x * blockDim.x + threadIdx.x;
  if (t >= MTOK) return;
  int b = t >> 10, n = t & 1023;
  float scl = fminf(fmaxf(orth_scale[0], 0.f), 1.f);
  float sacc = 0.f;
  for (int h = 0; h < HH; ++h) {
    size_t base = (((size_t)(b * HH + h)) * NSEQ + n) * HD;
    float dot = 0.f, nsq = 0.f;
    for (int d = 0; d < HD; ++d) {
      float a = (float)Qid[base + d];
      float c = (float)Qcl[base + d];
      dot += a * c; nsq += c * c;
    }
    nsq += 1e-5f;
    float coeff = fminf(fmaxf(dot / nsq, -1.f), 1.f);
    float f = scl * coeff;
    for (int d = 0; d < HD; ++d) {
      float a = (float)Qid[base + d];
      float c = (float)Qcl[base + d];
      Qid[base + d] = (_Float16)(a - f * c);
    }
    sacc += fabsf(dot) * rsqrtf(nsq);
  }
  sal[t] = fminf(fmaxf(sacc * (1.f / HH), 0.f), 1.f);
}

// ---------------------------------------------------------------- flash attention, one wave = 16 query rows
// Q,K in [B,H,N,D]; V pre-transposed to [B,H,D,N] -> all fragment loads are K-contiguous.
__global__ __launch_bounds__(128)
void opa_attn(const _Float16* __restrict__ Q, const _Float16* __restrict__ K,
              const _Float16* __restrict__ Vt, _Float16* __restrict__ O) {
  __shared__ _Float16 pstage[4][16 * 32];
  int wid  = threadIdx.x >> 5;
  int lane = threadIdx.x & 31;
  int bh   = blockIdx.y;
  int row0 = (blockIdx.x * 4 + wid) * 16;
  const _Float16* Qh = Q  + (size_t)bh * NSEQ * HD;
  const _Float16* Kh = K  + (size_t)bh * NSEQ * HD;
  const _Float16* Vh = Vt + (size_t)bh * HD * NSEQ;   // [D][N]

  v16h qa0 = load_A16x32(Qh + (size_t)row0 * HD, HD);
  v16h qa1 = load_A16x32(Qh + (size_t)row0 * HD + 32, HD);

  float mrun[8], lrun[8];
  v8f o[4];
#pragma unroll
  for (int g = 0; g < 8; ++g) { mrun[g] = -1e30f; lrun[g] = 0.f; }
#pragma unroll
  for (int t = 0; t < 4; ++t) o[t] = vzero8();

  for (int j0 = 0; j0 < NSEQ; j0 += 32) {
    v8f s0 = vzero8(), s1 = vzero8();
    v16h kb;
    kb = load_Bt32x16(Kh + (size_t)j0 * HD, HD);              s0 = wmma16(qa0, kb, s0);
    kb = load_Bt32x16(Kh + (size_t)j0 * HD + 32, HD);         s0 = wmma16(qa1, kb, s0);
    kb = load_Bt32x16(Kh + (size_t)(j0 + 16) * HD, HD);       s1 = wmma16(qa0, kb, s1);
    kb = load_Bt32x16(Kh + (size_t)(j0 + 16) * HD + 32, HD);  s1 = wmma16(qa1, kb, s1);

    float p0[8], p1[8], alpha[8];
#pragma unroll
    for (int g = 0; g < 8; ++g) {
      float a = fminf(fmaxf(s0[g] * 0.125f, -20.f), 20.f);
      float b = fminf(fmaxf(s1[g] * 0.125f, -20.f), 20.f);
      float rmax = hmax16(fmaxf(a, b));
      float mnew = fmaxf(mrun[g], rmax);
      alpha[g] = __expf(mrun[g] - mnew);
      mrun[g]  = mnew;
      p0[g] = __expf(a - mnew);
      p1[g] = __expf(b - mnew);
      lrun[g] = lrun[g] * alpha[g] + hsum16(p0[g] + p1[g]);
    }
#pragma unroll
    for (int t = 0; t < 4; ++t)
#pragma unroll
      for (int g = 0; g < 8; ++g) o[t][g] *= alpha[g];

    // C-layout -> A-layout via LDS round-trip (per-wave private 16x32 staging)
    int prow0 = (lane & 16) ? 8 : 0;
    int pc = lane & 15;
#pragma unroll
    for (int g = 0; g < 8; ++g) {
      pstage[wid][(prow0 + g) * 32 + pc]      = (_Float16)p0[g];
      pstage[wid][(prow0 + g) * 32 + 16 + pc] = (_Float16)p1[g];
    }
    __syncthreads();
    v16h pa = load_A16x32(&pstage[wid][0], 32);
#pragma unroll
    for (int t = 0; t < 4; ++t) {
      // B tile: keys j0..j0+31 (K) x dims t*16..t*16+15 (N), K-contiguous in Vt
      v16h vb = load_Bt32x16(Vh + (size_t)(t * 16) * NSEQ + j0, NSEQ);
      o[t] = wmma16(pa, vb, o[t]);
    }
    __syncthreads();
  }

  int b = bh / HH, h = bh - b * HH;
#pragma unroll
  for (int g = 0; g < 8; ++g) {
    float l = lrun[g];
    float inv = (l > 0.f) ? 1.f / l : 0.f;   // softmax NaN guard
    int n = row0 + ((lane & 16) ? 8 : 0) + g;
#pragma unroll
    for (int t = 0; t < 4; ++t) {
      int d = h * HD + t * 16 + (lane & 15);
      O[((size_t)(b * NSEQ + n)) * C_ + d] = (_Float16)(o[t][g] * inv);
    }
  }
}

// ---------------------------------------------------------------- proj GEMM + bias -> f32
__global__ __launch_bounds__(256)
void opa_gemm_proj(const _Float16* __restrict__ A, const _Float16* __restrict__ W,
                   const float* __restrict__ bias, float* __restrict__ out) {
  int wid  = threadIdx.x >> 5;
  int lane = threadIdx.x & 31;
  int row0 = blockIdx.y * 64  + (wid >> 2) * 32;
  int col0 = blockIdx.x * 256 + (wid & 3) * 64;
  v8f acc[2][4];
#pragma unroll
  for (int ri = 0; ri < 2; ++ri)
#pragma unroll
    for (int ci = 0; ci < 4; ++ci) acc[ri][ci] = vzero8();
  for (int kk = 0; kk < C_; kk += 32) {
    v16h a0 = load_A16x32(A + (size_t)row0 * C_ + kk, C_);
    v16h a1 = load_A16x32(A + (size_t)(row0 + 16) * C_ + kk, C_);
    v16h b[4];
#pragma unroll
    for (int ci = 0; ci < 4; ++ci)
      b[ci] = load_Bt32x16(W + (size_t)(col0 + ci * 16) * C_ + kk, C_);
#pragma unroll
    for (int ci = 0; ci < 4; ++ci) {
      acc[0][ci] = wmma16(a0, b[ci], acc[0][ci]);
      acc[1][ci] = wmma16(a1, b[ci], acc[1][ci]);
    }
  }
  int colt = lane & 15;
  int rowh = (lane & 16) ? 8 : 0;
#pragma unroll
  for (int ri = 0; ri < 2; ++ri)
#pragma unroll
    for (int ci = 0; ci < 4; ++ci)
#pragma unroll
      for (int g = 0; g < 8; ++g) {
        int m = row0 + ri * 16 + rowh + g;
        int c = col0 + ci * 16 + colt;
        out[(size_t)m * C_ + c] = acc[ri][ci][g] + bias[c];
      }
}

// ---------------------------------------------------------------- layernorm, one wave per row
__global__ __launch_bounds__(256)
void opa_ln(const float* __restrict__ X, const float* __restrict__ w,
            const float* __restrict__ bias, float* __restrict__ out) {
  int wid = threadIdx.x >> 5, lane = threadIdx.x & 31;
  int row = blockIdx.x * 8 + wid;
  const float* xr = X + (size_t)row * C_;
  float s = 0.f, s2 = 0.f;
  for (int i = lane; i < C_; i += 32) { float v = xr[i]; s += v; s2 += v * v; }
#pragma unroll
  for (int d = 1; d < 32; d <<= 1) { s += __shfl_xor(s, d, 32); s2 += __shfl_xor(s2, d, 32); }
  float mean = s * (1.f / C_);
  float var  = s2 * (1.f / C_) - mean * mean;
  float rstd = rsqrtf(var + 1e-5f);
  float* orow = out + (size_t)row * C_;
  for (int i = lane; i < C_; i += 32)
    orow[i] = (xr[i] - mean) * rstd * w[i] + bias[i];
}

// ----------------------------------------------------------------
extern "C" void kernel_launch(void* const* d_in, const int* in_sizes, int n_in,
                              void* d_out, int out_size, void* d_ws, size_t ws_size,
                              hipStream_t stream) {
  (void)in_sizes; (void)n_in; (void)out_size; (void)ws_size;
  const float* x        = (const float*)d_in[0];
  const float* wqkv_id  = (const float*)d_in[1];
  const float* wqkv_cl  = (const float*)d_in[2];
  const float* wproj_id = (const float*)d_in[3];
  const float* bproj_id = (const float*)d_in[4];
  const float* wproj_cl = (const float*)d_in[5];
  const float* bproj_cl = (const float*)d_in[6];
  const float* ln_id_w  = (const float*)d_in[7];
  const float* ln_id_b  = (const float*)d_in[8];
  const float* ln_cl_w  = (const float*)d_in[9];
  const float* ln_cl_b  = (const float*)d_in[10];
  const float* oscale   = (const float*)d_in[11];

  float* out_id = (float*)d_out;
  float* out_cl = out_id + (size_t)MTOK * C_;
  float* sal    = out_cl + (size_t)MTOK * C_;

  // workspace carve-up (f16 region first, f32 region after; all counts even -> aligned)
  _Float16* w16 = (_Float16*)d_ws;
  size_t off = 0;
  _Float16* Xh   = w16 + off; off += (size_t)MTOK * C_;
  _Float16* Wid  = w16 + off; off += (size_t)C3 * C_;
  _Float16* Wcl  = w16 + off; off += (size_t)C3 * C_;
  _Float16* Wpid = w16 + off; off += (size_t)C_ * C_;
  _Float16* Wpcl = w16 + off; off += (size_t)C_ * C_;
  const size_t hsz = (size_t)BH * NSEQ * HD;
  _Float16* Qid  = w16 + off; off += hsz;
  _Float16* Kid  = w16 + off; off += hsz;
  _Float16* Vtid = w16 + off; off += hsz;   // [B,H,D,N]
  _Float16* Qcl  = w16 + off; off += hsz;
  _Float16* Kcl  = w16 + off; off += hsz;
  _Float16* Vtcl = w16 + off; off += hsz;   // [B,H,D,N]
  _Float16* Oid  = w16 + off; off += (size_t)MTOK * C_;
  _Float16* Ocl  = w16 + off; off += (size_t)MTOK * C_;
  float* Pid = (float*)(w16 + off);
  float* Pcl = Pid + (size_t)MTOK * C_;

  // 1) precision conversion
  opa_cvt_f16<<<1024, 256, 0, stream>>>(x,        Xh,   MTOK * C_);
  opa_cvt_f16<<<512,  256, 0, stream>>>(wqkv_id,  Wid,  C3 * C_);
  opa_cvt_f16<<<512,  256, 0, stream>>>(wqkv_cl,  Wcl,  C3 * C_);
  opa_cvt_f16<<<256,  256, 0, stream>>>(wproj_id, Wpid, C_ * C_);
  opa_cvt_f16<<<256,  256, 0, stream>>>(wproj_cl, Wpcl, C_ * C_);

  // 2) QKV projections (V stored transposed)
  dim3 gq(C3 / 256, MTOK / 64);   // (9, 128)
  opa_gemm_qkv<<<gq, 256, 0, stream>>>(Xh, Wid, Qid, Kid, Vtid);
  opa_gemm_qkv<<<gq, 256, 0, stream>>>(Xh, Wcl, Qcl, Kcl, Vtcl);

  // 3) orthogonal projection of q_id against q_cloth + saliency
  opa_ortho<<<MTOK / 256, 256, 0, stream>>>(Qid, Qcl, oscale, sal);

  // 4) attention (two streams)
  dim3 ga(NSEQ / 64, BH);   // 16 blocks of 4 waves x 16 rows, 96 heads
  opa_attn<<<ga, 128, 0, stream>>>(Qid, Kid, Vtid, Oid);
  opa_attn<<<ga, 128, 0, stream>>>(Qcl, Kcl, Vtcl, Ocl);

  // 5) output projections + bias
  dim3 gp(C_ / 256, MTOK / 64);   // (3, 128)
  opa_gemm_proj<<<gp, 256, 0, stream>>>(Oid, Wpid, bproj_id, Pid);
  opa_gemm_proj<<<gp, 256, 0, stream>>>(Ocl, Wpcl, bproj_cl, Pcl);

  // 6) layernorm -> final outputs
  opa_ln<<<MTOK / 8, 256, 0, stream>>>(Pid, ln_id_w, ln_id_b, out_id);
  opa_ln<<<MTOK / 8, 256, 0, stream>>>(Pcl, ln_cl_w, ln_cl_b, out_cl);
}
